// ChannelAttention_31138512896075
// MI455X (gfx1250) — matrix operand order repaired and checked by reading, not compiled, not run on hardware
//
#include <hip/hip_runtime.h>
#include <hip/hip_bf16.h>
#include <math.h>
#include <stdint.h>

// ---------------------------------------------------------------------------
// ChannelAttention on MI455X (gfx1250), bf16 WMMA pipeline:
//   1) proj_kernel    : q/k/v [16384x1024] x W^T [1024x512] -> bf16 qt/kt/vt
//   2) transpose_bf16 : async-to-LDS staged 32x32 tile transpose; produces
//                       ktT[b][e][s] and vT[b][s][e] so all GEMM operand
//                       loads are 16B-vectorized (no strided gathers)
//   3) attn_logits    : per b: (512 x 2048) x (2048 x 512) -> f32 logits
//   4) softmax_mask   : mask + row softmax over 4096 rows x 512
//   5) attn_out       : per b: (512 x 512) x (512 x 2048) -> f32 out
// GEMMs use v_wmma_f32_16x16x32_bf16 (wave32, f32 accumulate).
// Transpose uses global_load_async_to_lds_b128 + s_wait_asynccnt (ASYNCcnt).
// ---------------------------------------------------------------------------

#define S_LEN 2048
#define B_SZ  8
#define D_SZ  1024
#define DH_SZ 512
#define M1    (S_LEN * B_SZ)   // 16384 rows in projection GEMM

typedef __bf16 bf16_t;
typedef __attribute__((ext_vector_type(16))) __bf16 v16bf;
typedef __attribute__((ext_vector_type(8)))  __bf16 v8bf;
typedef __attribute__((ext_vector_type(8)))  float  v8f;
typedef __attribute__((ext_vector_type(4)))  float  v4f;

// ---- fragment loaders (layouts per CDNA5 ISA 7.12.2) ------------

// A fragment 16x32 bf16 from row-major f32 source (convert on load).
// lane<16: row=m0+lane, K = k0 + {0..7, 16..23}
// lane>=16: row=m0+lane-16, K = k0 + {8..15, 24..31}
__device__ __forceinline__ v16bf frag_a_from_f32(const float* __restrict__ base,
                                                 int lda, int m0, int k0, int lane) {
  int row   = m0 + (lane & 15);
  int khalf = (lane >> 4) << 3;          // 0 or 8
  const float* p = base + row * lda + k0 + khalf;
  v4f x0 = *(const v4f*)(p);
  v4f x1 = *(const v4f*)(p + 4);
  v4f x2 = *(const v4f*)(p + 16);
  v4f x3 = *(const v4f*)(p + 20);
  v16bf a;
#pragma unroll
  for (int i = 0; i < 4; ++i) {
    a[i]      = (__bf16)x0[i];
    a[4 + i]  = (__bf16)x1[i];
    a[8 + i]  = (__bf16)x2[i];
    a[12 + i] = (__bf16)x3[i];
  }
  return a;
}

// A fragment 16x32 from row-major bf16 source (two 16B vector loads).
__device__ __forceinline__ v16bf frag_a_from_bf16(const bf16_t* __restrict__ base,
                                                  int lda, int m0, int k0, int lane) {
  int row   = m0 + (lane & 15);
  int khalf = (lane >> 4) << 3;
  const bf16_t* p = base + row * lda + k0 + khalf;
  v8bf x0 = *(const v8bf*)(p);           // K + 0..7
  v8bf x1 = *(const v8bf*)(p + 16);      // K + 16..23
  v16bf a;
#pragma unroll
  for (int i = 0; i < 8; ++i) { a[i] = x0[i]; a[8 + i] = x1[i]; }
  return a;
}

// B fragment 32x16: B(k,n) = W[n*ldk + k] (f32, contiguous in k, convert).
// lane<16: n=n0+lane, K=k0+{0..7,16..23} ; lane>=16: n=n0+lane-16, K=k0+{8..15,24..31}
__device__ __forceinline__ v16bf frag_b_from_Wf32(const float* __restrict__ W,
                                                  int ldk, int k0, int n0, int lane) {
  int n  = n0 + (lane & 15);
  int kb = k0 + ((lane >> 4) << 3);      // 0 or 8
  const float* p = W + n * ldk + kb;
  v4f x0 = *(const v4f*)(p);
  v4f x1 = *(const v4f*)(p + 4);
  v4f x2 = *(const v4f*)(p + 16);
  v4f x3 = *(const v4f*)(p + 20);
  v16bf b;
#pragma unroll
  for (int i = 0; i < 4; ++i) {
    b[i]      = (__bf16)x0[i];
    b[4 + i]  = (__bf16)x1[i];
    b[8 + i]  = (__bf16)x2[i];
    b[12 + i] = (__bf16)x3[i];
  }
  return b;
}

// B fragment 32x16 from N-major bf16 source: B(k,n) = base[n*ldk + k]
// (contiguous in k -> two 16B vector loads; produced by transpose kernel).
__device__ __forceinline__ v16bf frag_b_bf16_nmajor(const bf16_t* __restrict__ base,
                                                    int ldk, int k0, int n0, int lane) {
  int n  = n0 + (lane & 15);
  int kb = k0 + ((lane >> 4) << 3);
  const bf16_t* p = base + n * ldk + kb;
  v8bf x0 = *(const v8bf*)(p);           // K kb+0..7
  v8bf x1 = *(const v8bf*)(p + 16);      // K kb+16..23
  v16bf b;
#pragma unroll
  for (int i = 0; i < 8; ++i) { b[i] = x0[i]; b[8 + i] = x1[i]; }
  return b;
}

#define WMMA_BF16(a, b, c) \
  __builtin_amdgcn_wmma_f32_16x16x32_bf16(false, (a), false, (b), (short)0, (c), false, false)

// ---------------------------------------------------------------------------
// Kernel 1: projections. grid (64, 8, 3), block 256 (8 waves).
// Wave tile 32(M) x 64(N); block tile 256 x 64. K = 1024.
// ---------------------------------------------------------------------------
__global__ __launch_bounds__(256) void proj_kernel(
    const float* __restrict__ q, const float* __restrict__ k, const float* __restrict__ v,
    const float* __restrict__ Wq, const float* __restrict__ bq,
    const float* __restrict__ Wk, const float* __restrict__ bk,
    const float* __restrict__ Wv, const float* __restrict__ bv,
    bf16_t* __restrict__ qt, bf16_t* __restrict__ kt, bf16_t* __restrict__ vt) {
  int which = blockIdx.z;
  const float* A    = (which == 0) ? q  : (which == 1) ? k  : v;
  const float* W    = (which == 0) ? Wq : (which == 1) ? Wk : Wv;
  const float* bias = (which == 0) ? bq : (which == 1) ? bk : bv;
  bf16_t*      out  = (which == 0) ? qt : (which == 1) ? kt : vt;

  int wave = threadIdx.x >> 5;
  int lane = threadIdx.x & 31;
  int m0 = blockIdx.x * 256 + wave * 32;
  int n0 = blockIdx.y * 64;

  v8f acc[2][4] = {};
  for (int k0 = 0; k0 < D_SZ; k0 += 32) {
    if (k0 + 32 < D_SZ) {
      // stream hint for the next A tile (emits global_prefetch_b8)
      __builtin_prefetch(A + (m0 + (lane & 15)) * D_SZ + k0 + 32, 0, 0);
    }
    v16bf a0 = frag_a_from_f32(A, D_SZ, m0,      k0, lane);
    v16bf a1 = frag_a_from_f32(A, D_SZ, m0 + 16, k0, lane);
#pragma unroll
    for (int j = 0; j < 4; ++j) {
      v16bf bf = frag_b_from_Wf32(W, D_SZ, k0, n0 + 16 * j, lane);
      acc[0][j] = WMMA_BF16(a0, bf, acc[0][j]);
      acc[1][j] = WMMA_BF16(a1, bf, acc[1][j]);
    }
  }

  int cl = lane & 15;
  int rhalf = (lane >> 4) << 3;
#pragma unroll
  for (int h = 0; h < 2; ++h) {
    int rbase = m0 + 16 * h + rhalf;
#pragma unroll
    for (int j = 0; j < 4; ++j) {
      int col = n0 + 16 * j + cl;
      float bb = bias[col];
#pragma unroll
      for (int r = 0; r < 8; ++r) {
        out[(rbase + r) * DH_SZ + col] = (bf16_t)(acc[h][j][r] + bb);
      }
    }
  }
}

// ---------------------------------------------------------------------------
// Kernel 2: batched bf16 matrix transpose via async-to-LDS staging.
// Per batch b: src (R x C row-major) -> dst (C x R row-major), dst[c][r]=src[r][c].
// grid (C/32, R/32, 8), block 128. Tile 32x32.
// Staging: 128 lanes each issue one global_load_async_to_lds_b128 (16B),
// s_wait_asynccnt 0, barrier, then transposed LDS reads + coalesced stores.
// ---------------------------------------------------------------------------
__global__ __launch_bounds__(128) void transpose_bf16_kernel(
    const bf16_t* __restrict__ src, bf16_t* __restrict__ dst, int R, int C) {
  int b = blockIdx.z;
  src += (size_t)b * R * C;
  dst += (size_t)b * R * C;
  int r0 = blockIdx.y * 32;
  int c0 = blockIdx.x * 32;

  __shared__ bf16_t sh[32][40];          // 80B row stride: 16B aligned, bank-spread

  int t  = threadIdx.x;
  int tr = t >> 2;                       // 0..31 : tile row
  int tc = (t & 3) * 8;                  // 0,8,16,24 : col group (8 bf16 = 16B)

  {
    unsigned int lds_addr = (unsigned int)(uintptr_t)&sh[tr][tc];
    unsigned long long ga =
        (unsigned long long)(uintptr_t)(src + (size_t)(r0 + tr) * C + (c0 + tc));
    asm volatile("global_load_async_to_lds_b128 %0, %1, off"
                 :: "v"(lds_addr), "v"(ga) : "memory");
  }
  asm volatile("s_wait_asynccnt 0x0" ::: "memory");
  __syncthreads();

  // out row c0+tr, cols r0+tc .. +7 : dst[c][r] = sh[r - r0][c - c0]
  v8bf o;
#pragma unroll
  for (int i = 0; i < 8; ++i) o[i] = sh[tc + i][tr];
  *(v8bf*)(dst + (size_t)(c0 + tr) * R + (r0 + tc)) = o;
}

// ---------------------------------------------------------------------------
// Kernel 3: logits[b,d,e] = sum_s q_[b,d,s] * k_[b,s,e]
//  A: qt flat + b*DH*S, row-major (d, s), lda = S
//  B: ktT + b*DH*S, N-major (e, s), ldk = S  (from transpose kernel)
// grid (2, 16, 8), block 128 (4 waves). Wave tile 64(M) x 32(N), K = 2048.
// ---------------------------------------------------------------------------
__global__ __launch_bounds__(128) void attn_logits_kernel(
    const bf16_t* __restrict__ qt, const bf16_t* __restrict__ ktT,
    float* __restrict__ attn) {
  int wave = threadIdx.x >> 5;
  int lane = threadIdx.x & 31;
  int b  = blockIdx.z;
  int d0 = blockIdx.x * 256 + wave * 64;
  int e0 = blockIdx.y * 32;
  const bf16_t* Qb = qt  + b * (DH_SZ * S_LEN);
  const bf16_t* Bb = ktT + b * (DH_SZ * S_LEN);

  v8f acc[4][2] = {};
  for (int k0 = 0; k0 < S_LEN; k0 += 32) {
    v16bf bf0 = frag_b_bf16_nmajor(Bb, S_LEN, k0, e0,      lane);
    v16bf bf1 = frag_b_bf16_nmajor(Bb, S_LEN, k0, e0 + 16, lane);
#pragma unroll
    for (int i = 0; i < 4; ++i) {
      v16bf a = frag_a_from_bf16(Qb, S_LEN, d0 + 16 * i, k0, lane);
      acc[i][0] = WMMA_BF16(a, bf0, acc[i][0]);
      acc[i][1] = WMMA_BF16(a, bf1, acc[i][1]);
    }
  }

  float* Ab = attn + b * (DH_SZ * DH_SZ);
  int rhalf = (lane >> 4) << 3;
  int cl = lane & 15;
#pragma unroll
  for (int i = 0; i < 4; ++i)
#pragma unroll
    for (int j = 0; j < 2; ++j)
#pragma unroll
      for (int r = 0; r < 8; ++r)
        Ab[(d0 + 16 * i + rhalf + r) * DH_SZ + (e0 + 16 * j + cl)] = acc[i][j][r];
}

// ---------------------------------------------------------------------------
// Kernel 4: masked softmax over last axis. One 256-thread block per (b,d) row.
// grid 4096, block 256. mask[d,e] true -> -inf before softmax.
// ---------------------------------------------------------------------------
__global__ __launch_bounds__(256) void softmax_mask_kernel(
    float* __restrict__ attn, const unsigned char* __restrict__ mask) {
  int row = blockIdx.x;                 // b*DH + d
  int d = row & (DH_SZ - 1);
  float* p = attn + (size_t)row * DH_SZ;
  const unsigned char* m = mask + (size_t)d * DH_SZ;
  int t = threadIdx.x;

  float x0 = m[t]       ? -INFINITY : p[t];
  float x1 = m[t + 256] ? -INFINITY : p[t + 256];

  __shared__ float redmax[8];
  __shared__ float redsum[8];

  float mx = fmaxf(x0, x1);
#pragma unroll
  for (int off = 16; off > 0; off >>= 1) mx = fmaxf(mx, __shfl_xor(mx, off, 32));
  if ((t & 31) == 0) redmax[t >> 5] = mx;
  __syncthreads();
  mx = redmax[0];
#pragma unroll
  for (int i = 1; i < 8; ++i) mx = fmaxf(mx, redmax[i]);

  float e0 = __expf(x0 - mx);
  float e1 = __expf(x1 - mx);
  float sm = e0 + e1;
#pragma unroll
  for (int off = 16; off > 0; off >>= 1) sm += __shfl_xor(sm, off, 32);
  if ((t & 31) == 0) redsum[t >> 5] = sm;
  __syncthreads();
  sm = redsum[0];
#pragma unroll
  for (int i = 1; i < 8; ++i) sm += redsum[i];

  float inv = 1.0f / sm;
  p[t]       = e0 * inv;
  p[t + 256] = e1 * inv;
}

// ---------------------------------------------------------------------------
// Kernel 5: out[b,d,s] = sum_e attn[b,d,e] * v_[b,e,s]
//  A: attn f32 row-major (d,e), lda = DH (convert to bf16 on load)
//  B: vT + b*S*DH, N-major (s, e), ldk = DH  (from transpose kernel)
//  out written at flat b*DH*S + d*S + s (== (S,B,DH) reference output order)
// grid (2, 64, 8), block 128 (4 waves). Wave tile 64(M) x 32(N), K = 512.
// ---------------------------------------------------------------------------
__global__ __launch_bounds__(128) void attn_out_kernel(
    const float* __restrict__ attn, const bf16_t* __restrict__ vT,
    float* __restrict__ out) {
  int wave = threadIdx.x >> 5;
  int lane = threadIdx.x & 31;
  int b  = blockIdx.z;
  int d0 = blockIdx.x * 256 + wave * 64;
  int s0 = blockIdx.y * 32;
  const float*  Ab = attn + b * (DH_SZ * DH_SZ);
  const bf16_t* Bb = vT + b * (S_LEN * DH_SZ);

  v8f acc[4][2] = {};
  for (int k0 = 0; k0 < DH_SZ; k0 += 32) {
    v16bf bf0 = frag_b_bf16_nmajor(Bb, DH_SZ, k0, s0,      lane);
    v16bf bf1 = frag_b_bf16_nmajor(Bb, DH_SZ, k0, s0 + 16, lane);
#pragma unroll
    for (int i = 0; i < 4; ++i) {
      v16bf a = frag_a_from_f32(Ab, DH_SZ, d0 + 16 * i, k0, lane);
      acc[i][0] = WMMA_BF16(a, bf0, acc[i][0]);
      acc[i][1] = WMMA_BF16(a, bf1, acc[i][1]);
    }
  }

  float* Ob = out + b * (DH_SZ * S_LEN);
  int rhalf = (lane >> 4) << 3;
  int cl = lane & 15;
#pragma unroll
  for (int i = 0; i < 4; ++i)
#pragma unroll
    for (int j = 0; j < 2; ++j)
#pragma unroll
      for (int r = 0; r < 8; ++r)
        Ob[(d0 + 16 * i + rhalf + r) * S_LEN + (s0 + 16 * j + cl)] = acc[i][j][r];
}

// ---------------------------------------------------------------------------
extern "C" void kernel_launch(void* const* d_in, const int* in_sizes, int n_in,
                              void* d_out, int out_size, void* d_ws, size_t ws_size,
                              hipStream_t stream) {
  const float* q  = (const float*)d_in[0];
  const float* k  = (const float*)d_in[1];
  const float* v  = (const float*)d_in[2];
  const unsigned char* mask = (const unsigned char*)d_in[3];  // bool (DH,DH)
  const float* Wq = (const float*)d_in[4];
  const float* bq = (const float*)d_in[5];
  const float* Wk = (const float*)d_in[6];
  const float* bk = (const float*)d_in[7];
  const float* Wv = (const float*)d_in[8];
  const float* bv = (const float*)d_in[9];
  float* out = (float*)d_out;

  // workspace layout
  const size_t PROJ_ELEMS = (size_t)M1 * DH_SZ;                 // 8,388,608
  char* ws = (char*)d_ws;
  bf16_t* qt  = (bf16_t*)(ws);
  bf16_t* kt  = (bf16_t*)(ws + 1 * PROJ_ELEMS * sizeof(bf16_t));
  bf16_t* vt  = (bf16_t*)(ws + 2 * PROJ_ELEMS * sizeof(bf16_t));
  bf16_t* ktT = (bf16_t*)(ws + 3 * PROJ_ELEMS * sizeof(bf16_t));
  bf16_t* vT  = (bf16_t*)(ws + 4 * PROJ_ELEMS * sizeof(bf16_t));
  float*  att = (float*) (ws + 5 * PROJ_ELEMS * sizeof(bf16_t));
  // total: 5*16MB bf16 + 8MB attn = ~92.3 MB

  proj_kernel<<<dim3(M1 / 256, DH_SZ / 64, 3), 256, 0, stream>>>(
      q, k, v, Wq, bq, Wk, bk, Wv, bv, qt, kt, vt);

  // ktT[b][e][s] = k_[b,s,e] : per-b transpose of (S x DH) -> (DH x S)
  transpose_bf16_kernel<<<dim3(DH_SZ / 32, S_LEN / 32, B_SZ), 128, 0, stream>>>(
      kt, ktT, S_LEN, DH_SZ);
  // vT[b][s][e] = v_[b,e,s] : per-b transpose of (DH x S) -> (S x DH)
  transpose_bf16_kernel<<<dim3(S_LEN / 32, DH_SZ / 32, B_SZ), 128, 0, stream>>>(
      vt, vT, DH_SZ, S_LEN);

  attn_logits_kernel<<<dim3(2, 16, B_SZ), 128, 0, stream>>>(qt, ktT, att);

  softmax_mask_kernel<<<dim3(B_SZ * DH_SZ), 256, 0, stream>>>(att, mask);

  attn_out_kernel<<<dim3(2, 64, B_SZ), 128, 0, stream>>>(att, vT, out);
}
